// IASSD_Backbone_2113123910213
// MI455X (gfx1250) — compile-verified, matrix operand order checked by reading, not run on hardware
//
#include <hip/hip_runtime.h>
#include <stdint.h>

#define BATCH 8
#define NP    2048
#define NC    1024
#define PW    (NP / 32)   // 64 bitmask words per row

typedef __attribute__((ext_vector_type(16))) _Float16 v16h;
typedef __attribute__((ext_vector_type(8)))  float    v8f;

// per-point stats: squared norm, argmax label (first-max tie-break), radius^2
__device__ __forceinline__ void point_stats(const float* __restrict__ centers,
                                            const float* __restrict__ cls,
                                            const float* __restrict__ radius,
                                            int bp, float& nrm, int& lab, float& r2)
{
    float x = centers[bp * 3 + 0];
    float y = centers[bp * 3 + 1];
    float z = centers[bp * 3 + 2];
    nrm = x * x + y * y + z * z;
    float s0 = cls[bp * 3 + 0];
    float s1 = cls[bp * 3 + 1];
    float s2 = cls[bp * 3 + 2];
    int   l = 0; float bb = s0;
    if (s1 > bb) { bb = s1; l = 1; }
    if (s2 > bb) { bb = s2; l = 2; }
    lab = l;
    float r = radius[l];
    r2 = r * r;
}

// ---------------------------------------------------------------------------
// Kernel 1: build bit-packed suppress matrix with WMMA Gram tiles.
// Block = 256 threads = 8 waves; wave w handles i-tile (blockIdx.y*8 + w),
// block handles 32 consecutive j-columns (two 16-wide WMMA B tiles), so each
// wave owns full u32 bitmask words -> no write races.
// suppress[b][i][j] = (label[i]==label[j]) && dist(i,j) < radius[label[i]]
// ---------------------------------------------------------------------------
__global__ __launch_bounds__(256)
void suppress_build(const float* __restrict__ centers,
                    const float* __restrict__ cls,
                    const float* __restrict__ radius,
                    uint32_t* __restrict__ suppressG)
{
    const int lane  = threadIdx.x & 31;
    const int wave  = threadIdx.x >> 5;
    const int b     = blockIdx.z;
    const int itile = blockIdx.y * 8 + wave;   // 0..127
    const int jpair = blockIdx.x;              // 0..63  (32 columns)
    const int ibase = itile * 16;
    const int jbase = jpair * 32;
    const int bp0   = b * NP;

    // A: 16x32 f16, rows in lanes 0..15, K=0..2 = xyz.  B: 32x16 f16,
    // cols in lanes 0..15 (K=0..15 half-indexed), K=0..2 = xyz.
    v16h a  = {};
    v16h bm0 = {};
    v16h bm1 = {};
    if (lane < 16) {
        int ri = bp0 + ibase + lane;
        a[0] = (_Float16)centers[ri * 3 + 0];
        a[1] = (_Float16)centers[ri * 3 + 1];
        a[2] = (_Float16)centers[ri * 3 + 2];
        int j0 = bp0 + jbase + lane;
        bm0[0] = (_Float16)centers[j0 * 3 + 0];
        bm0[1] = (_Float16)centers[j0 * 3 + 1];
        bm0[2] = (_Float16)centers[j0 * 3 + 2];
        int j1 = j0 + 16;
        bm1[0] = (_Float16)centers[j1 * 3 + 0];
        bm1[1] = (_Float16)centers[j1 * 3 + 1];
        bm1[2] = (_Float16)centers[j1 * 3 + 2];
    }
    v8f c0 = {};
    v8f c1 = {};
    c0 = __builtin_amdgcn_wmma_f32_16x16x32_f16(false, a, false, bm0, (short)0, c0, false, false);
    c1 = __builtin_amdgcn_wmma_f32_16x16x32_f16(false, a, false, bm1, (short)0, c1, false, false);

    // per-lane column stats (N = lane%16), duplicated across half-waves
    const int n = lane & 15;
    float cn0, cn1, dum;
    int   cl0, cl1, dl;
    point_stats(centers, cls, radius, bp0 + jbase + n,      cn0, cl0, dum);
    point_stats(centers, cls, radius, bp0 + jbase + 16 + n, cn1, cl1, dum);
    (void)dl;

    // per-lane row stats for the 8 rows this half-wave owns in C/D
    const int mhalf = bp0 + ibase + ((lane >> 4) << 3);
    float rn[8], rr2[8];
    int   rl[8];
#pragma unroll
    for (int r = 0; r < 8; ++r)
        point_stats(centers, cls, radius, mhalf + r, rn[r], rl[r], rr2[r]);

#pragma unroll
    for (int r = 0; r < 8; ++r) {
        // D[m][n] = dot(center_m, center_n); dist^2 = |m|^2 + |n|^2 - 2 dot
        float d20 = rn[r] + cn0 - 2.0f * c0[r];
        float d21 = rn[r] + cn1 - 2.0f * c1[r];
        bool p0 = (rl[r] == cl0) && (d20 < rr2[r]);
        bool p1 = (rl[r] == cl1) && (d21 < rr2[r]);
        uint32_t m0 = (uint32_t)__ballot((int)p0);   // bits 0-15: row r, 16-31: row r+8
        uint32_t m1 = (uint32_t)__ballot((int)p1);
        if (lane == 0) {
            uint32_t wlo = (m0 & 0xFFFFu) | (m1 << 16);          // row r,   cols jbase..+31
            uint32_t whi = (m0 >> 16)    | (m1 & 0xFFFF0000u);   // row r+8
            suppressG[(size_t)(bp0 + ibase + r)     * PW + jpair] = wlo;
            suppressG[(size_t)(bp0 + ibase + 8 + r) * PW + jpair] = whi;
        }
    }
}

// ---------------------------------------------------------------------------
// Kernel 2: greedy score-ordered NMS, one scene per 1024-thread block.
// ---------------------------------------------------------------------------
__global__ __launch_bounds__(1024)
void greedy_nms(const float* __restrict__ centers,
                const float* __restrict__ cls,
                const uint32_t* __restrict__ suppressG,
                uint32_t* __restrict__ keepG,
                float* __restrict__ out_centers,
                float* __restrict__ out_cls,
                float* __restrict__ out_keep)
{
    __shared__ float    s_scores[NP];
    __shared__ uint32_t s_alive[PW];
    __shared__ uint32_t s_keep[PW];
    __shared__ unsigned long long s_red[32];
    __shared__ unsigned long long s_wkey;

    const int b   = blockIdx.x;
    const int tid = threadIdx.x;
    const int bp0 = b * NP;

#pragma unroll
    for (int h = 0; h < 2; ++h) {
        int p = tid + h * 1024;
        float s0 = cls[(size_t)(bp0 + p) * 3 + 0];
        float s1 = cls[(size_t)(bp0 + p) * 3 + 1];
        float s2 = cls[(size_t)(bp0 + p) * 3 + 2];
        s_scores[p] = fmaxf(s0, fmaxf(s1, s2));
    }
    if (tid < PW) { s_alive[tid] = 0xFFFFFFFFu; s_keep[tid] = 0u; }
    __syncthreads();

    for (int iter = 0; iter < NP; ++iter) {
        // ---- argmax over alive points: key = (ordered_f32 << 32) | ~idx
        unsigned long long best = 0ull;
#pragma unroll
        for (int h = 0; h < 2; ++h) {
            int p = tid + h * 1024;
            if ((s_alive[p >> 5] >> (p & 31)) & 1u) {
                uint32_t u = __float_as_uint(s_scores[p]);
                u = (u & 0x80000000u) ? ~u : (u | 0x80000000u);
                unsigned long long key =
                    ((unsigned long long)u << 32) | (uint32_t)(~(uint32_t)p);
                if (key > best) best = key;
            }
        }
#pragma unroll
        for (int off = 16; off > 0; off >>= 1) {
            unsigned long long o = __shfl_down(best, off, 32);
            if (o > best) best = o;
        }
        if ((tid & 31) == 0) s_red[tid >> 5] = best;
        __syncthreads();
        if (tid < 32) {
            unsigned long long v = s_red[tid];
#pragma unroll
            for (int off = 16; off > 0; off >>= 1) {
                unsigned long long o = __shfl_down(v, off, 32);
                if (o > v) v = o;
            }
            if (tid == 0) s_wkey = v;
        }
        __syncthreads();
        unsigned long long wk = s_wkey;
        if (wk == 0ull) break;                      // nothing alive (uniform)
        int w = (int)(~(uint32_t)wk) & (NP - 1);    // winner index

        if (tid == 0) s_keep[w >> 5] |= 1u << (w & 31);
        if (tid < PW) {
            uint32_t sup = suppressG[(size_t)(bp0 + w) * PW + tid];
            if (tid == (w >> 5)) sup |= 1u << (w & 31);   // always retire winner
            s_alive[tid] &= ~sup;
        }
        __syncthreads();
    }

    if (tid < PW) keepG[b * PW + tid] = s_keep[tid];
    __syncthreads();

    // small masked outputs + keep
#pragma unroll
    for (int h = 0; h < 2; ++h) {
        int p = tid + h * 1024;
        float m = ((s_keep[p >> 5] >> (p & 31)) & 1u) ? 1.0f : 0.0f;
        out_keep[bp0 + p] = m;
        size_t base = (size_t)(bp0 + p) * 3;
        out_centers[base + 0] = centers[base + 0] * m;
        out_centers[base + 1] = centers[base + 1] * m;
        out_centers[base + 2] = centers[base + 2] * m;
        out_cls[base + 0] = cls[base + 0] * m;
        out_cls[base + 1] = cls[base + 1] * m;
        out_cls[base + 2] = cls[base + 2] * m;
    }
}

// ---------------------------------------------------------------------------
// Kernel 3: bandwidth-dominant features masking, float4 vectorized.
// ---------------------------------------------------------------------------
__global__ __launch_bounds__(256)
void mask_features(const float4* __restrict__ fin,
                   const uint32_t* __restrict__ keepG,
                   float4* __restrict__ fout)
{
    int i  = blockIdx.x * blockDim.x + threadIdx.x;   // 0 .. B*P*C/4 - 1
    int pg = i >> 8;                                  // 256 float4 per point (C=1024)
    float m = ((keepG[pg >> 5] >> (pg & 31)) & 1u) ? 1.0f : 0.0f;
    float4 v = fin[i];
    v.x *= m; v.y *= m; v.z *= m; v.w *= m;
    fout[i] = v;
}

// ---------------------------------------------------------------------------
extern "C" void kernel_launch(void* const* d_in, const int* in_sizes, int n_in,
                              void* d_out, int out_size, void* d_ws, size_t ws_size,
                              hipStream_t stream)
{
    const float* centers  = (const float*)d_in[0];   // [8,2048,3]
    const float* features = (const float*)d_in[1];   // [8,2048,1024]
    const float* cls      = (const float*)d_in[2];   // [8,2048,3]
    const float* radius   = (const float*)d_in[3];   // [3]

    float* out_centers  = (float*)d_out;
    float* out_features = out_centers  + (size_t)BATCH * NP * 3;
    float* out_cls      = out_features + (size_t)BATCH * NP * NC;
    float* out_keep     = out_cls      + (size_t)BATCH * NP * 3;

    uint32_t* suppressG = (uint32_t*)d_ws;                       // 8*2048*64 u32 = 4 MB
    uint32_t* keepG     = suppressG + (size_t)BATCH * NP * PW;   // 8*64 u32

    suppress_build<<<dim3(64, 16, BATCH), 256, 0, stream>>>(centers, cls, radius, suppressG);
    greedy_nms<<<dim3(BATCH), 1024, 0, stream>>>(centers, cls, suppressG, keepG,
                                                 out_centers, out_cls, out_keep);
    int f4 = (BATCH * NP * NC) / 4;
    mask_features<<<dim3(f4 / 256), 256, 0, stream>>>((const float4*)features, keepG,
                                                      (float4*)out_features);
}
